// Block_72121090834687
// MI455X (gfx1250) — matrix operand order repaired and checked by reading, not compiled, not run on hardware
//
#include <hip/hip_runtime.h>
#include <hip/hip_bf16.h>

// ---------------------------------------------------------------------------
// CDNA5 (gfx1250) implementation: all GEMMs via v_wmma_f32_16x16x32_bf16,
// fp32 LayerNorm/softmax/accumulation, bf16 matmul operands.
// Round 3: ping-pong double buffering (k-loop unrolled x2) removes the
// v_dual_mov buffer-rotation copies seen in round 2's inner loop.
// ---------------------------------------------------------------------------

typedef __bf16 bf16x8 __attribute__((ext_vector_type(8)));
typedef __bf16 v16bf  __attribute__((ext_vector_type(16)));
typedef float  v8f    __attribute__((ext_vector_type(8)));

#define BATCH   8
#define CDIM    256
#define NTOK    3600
#define PTOK    1769
#define KTVAL   2025
#define KTPAD   2048
#define NHEAD   8
#define DHEAD   32
#define DFF     1024

__device__ inline v16bf ld16(const __bf16* plo, const __bf16* phi) {
  bf16x8 lo = *(const bf16x8*)plo;
  bf16x8 hi = *(const bf16x8*)phi;
  v16bf r;
#pragma unroll
  for (int i = 0; i < 8; i++) { r[i] = lo[i]; r[i + 8] = hi[i]; }
  return r;
}

__device__ inline v8f wmma_bf16(v16bf a, v16bf b, v8f c) {
  return __builtin_amdgcn_wmma_f32_16x16x32_bf16(false, a, false, b, (short)0, c,
                                                 false, false);
}

__device__ inline float gelu_exact(float v) {
  return 0.5f * v * (1.f + erff(v * 0.70710678118654752f));
}

// ---------------------------------------------------------------------------
// Generic NT GEMM:  C[M,N] = A[M,K] * Bt[N,K]^T  (+bias, +res, gelu, bf16 out)
// block = 128 threads (4 waves); block tile 64x64; wave tile 16x64.
// k-loop unrolled x2 with ping-pong fragment buffers: loads for step i+1
// issue before the WMMAs of step i, with no buffer-swap register moves.
// ---------------------------------------------------------------------------
__global__ __launch_bounds__(128) void gemm_nt_wmma(
    const __bf16* __restrict__ A, long sAb, long sAh, int lda,
    const __bf16* __restrict__ Bt, long sBb, long sBh, int ldb,
    float* __restrict__ Cf, __bf16* __restrict__ Cb, long sCb, long sCh, int ldc,
    const float* __restrict__ bias, const float* __restrict__ res,
    int M, int N, int K, int nbh, int doGelu) {
  const int z = blockIdx.z;
  const int bi = z / nbh, hi = z - bi * nbh;
  A  += (long)bi * sAb + (long)hi * sAh;
  Bt += (long)bi * sBb + (long)hi * sBh;
  const long cbase = (long)bi * sCb + (long)hi * sCh;

  const int wv = threadIdx.x >> 5;
  const int lane = threadIdx.x & 31;
  const int lh = lane >> 4, ll = lane & 15;
  const int rowBase = blockIdx.y * 64 + wv * 16;
  const int colBase = blockIdx.x * 64;

  v8f acc[4];
#pragma unroll
  for (int t = 0; t < 4; t++)
#pragma unroll
    for (int j = 0; j < 8; j++) acc[t][j] = 0.f;

  int aRow = rowBase + ll; if (aRow > M - 1) aRow = M - 1;
  const __bf16* ap = A + (long)aRow * lda;
  const __bf16* bp[4];
#pragma unroll
  for (int t = 0; t < 4; t++) {
    int n = colBase + t * 16 + ll; if (n > N - 1) n = N - 1;
    bp[t] = Bt + (long)n * ldb;
  }

  v16bf a0, a1, b0[4], b1[4];

  // prologue: buffer 0 <- k = 0
  a0 = ld16(ap + lh * 16, ap + lh * 16 + 8);
#pragma unroll
  for (int t = 0; t < 4; t++)
    b0[t] = ld16(bp[t] + lh * 16, bp[t] + lh * 16 + 8);

  int k0 = 0;
  for (; k0 + 32 < K; k0 += 64) {
    // buffer 1 <- k0 + 32
    int ka = k0 + 32;
    a1 = ld16(ap + ka + lh * 16, ap + ka + lh * 16 + 8);
#pragma unroll
    for (int t = 0; t < 4; t++)
      b1[t] = ld16(bp[t] + ka + lh * 16, bp[t] + ka + lh * 16 + 8);
    // consume buffer 0 (k0)
#pragma unroll
    for (int t = 0; t < 4; t++) acc[t] = wmma_bf16(a0, b0[t], acc[t]);
    // buffer 0 <- k0 + 64 (clamped: final pass redundantly reloads K-32)
    int kb = k0 + 64; if (kb > K - 32) kb = K - 32;
    a0 = ld16(ap + kb + lh * 16, ap + kb + lh * 16 + 8);
#pragma unroll
    for (int t = 0; t < 4; t++)
      b0[t] = ld16(bp[t] + kb + lh * 16, bp[t] + kb + lh * 16 + 8);
    // consume buffer 1 (k0 + 32)
#pragma unroll
    for (int t = 0; t < 4; t++) acc[t] = wmma_bf16(a1, b1[t], acc[t]);
  }
  if (k0 < K) {  // odd number of 32-steps (e.g. K == 32)
#pragma unroll
    for (int t = 0; t < 4; t++) acc[t] = wmma_bf16(a0, b0[t], acc[t]);
  }

#pragma unroll
  for (int t = 0; t < 4; t++) {
    int col = colBase + t * 16 + ll;
    if (col < N) {
      float badd = bias ? bias[col] : 0.f;
#pragma unroll
      for (int j = 0; j < 8; j++) {
        int row = rowBase + lh * 8 + j;
        if (row < M) {
          float v = acc[t][j] + badd;
          long idx = cbase + (long)row * ldc + col;
          if (res) v += res[idx];
          if (doGelu) v = gelu_exact(v);
          if (Cf) Cf[idx] = v;
          if (Cb) Cb[idx] = (__bf16)v;
        }
      }
    }
  }
}

// ---------------------------------------------------------------------------
// LayerNorm over C=256 (fp32 in, bf16 out). One wave per row.
// ---------------------------------------------------------------------------
__global__ __launch_bounds__(128) void layernorm_bf16(
    const float* __restrict__ in, const float* __restrict__ g,
    const float* __restrict__ be, __bf16* __restrict__ out,
    int totalRows, int rowsPerBatch, long outBatchStride, long outOff) {
  int row = blockIdx.x * 4 + (threadIdx.x >> 5);
  if (row >= totalRows) return;
  int lane = threadIdx.x & 31;
  const float* p = in + (long)row * CDIM + lane * 8;
  float x[8];
#pragma unroll
  for (int i = 0; i < 8; i++) x[i] = p[i];
  float s = 0.f, s2 = 0.f;
#pragma unroll
  for (int i = 0; i < 8; i++) { s += x[i]; s2 += x[i] * x[i]; }
#pragma unroll
  for (int o = 1; o < 32; o <<= 1) {
    s += __shfl_xor(s, o, 32);
    s2 += __shfl_xor(s2, o, 32);
  }
  float mean = s * (1.f / CDIM);
  float var = s2 * (1.f / CDIM) - mean * mean;
  float rstd = rsqrtf(var + 1e-5f);
  int bidx = row / rowsPerBatch;
  int r = row - bidx * rowsPerBatch;
  __bf16* o = out + (long)bidx * outBatchStride + outOff + (long)r * CDIM + lane * 8;
  bf16x8 ov;
#pragma unroll
  for (int i = 0; i < 8; i++) {
    int c = lane * 8 + i;
    ov[i] = (__bf16)((x[i] - mean) * rstd * g[c] + be[c]);
  }
  *(bf16x8*)o = ov;
}

// ---------------------------------------------------------------------------
// Multi-scale pooling of xn (bf16 NHWC): one block per output token.
// Regions: avg 20x20(k3), 15x15(k4), 12x12(k5), 10x10(k6), max 30x30(k2).
// ---------------------------------------------------------------------------
__global__ __launch_bounds__(256) void pool_kernel(
    const __bf16* __restrict__ xn, float* __restrict__ ptf,
    __bf16* __restrict__ ptb) {
  int tk = blockIdx.x % PTOK;
  int b = blockIdx.x / PTOK;
  int c = threadIdx.x;
  int off, o, k; bool isMax;
  if (tk < 400)      { off = 0;   o = 20; k = 3; isMax = false; }
  else if (tk < 625) { off = 400; o = 15; k = 4; isMax = false; }
  else if (tk < 769) { off = 625; o = 12; k = 5; isMax = false; }
  else if (tk < 869) { off = 769; o = 10; k = 6; isMax = false; }
  else               { off = 869; o = 30; k = 2; isMax = true;  }
  int local = tk - off;
  int oy = local / o, ox = local - oy * o;
  const __bf16* base = xn + (long)b * NTOK * CDIM + c;
  float acc = isMax ? -1e30f : 0.f;
  for (int dy = 0; dy < k; dy++)
    for (int dx = 0; dx < k; dx++) {
      float v = (float)base[(long)((oy * k + dy) * 60 + (ox * k + dx)) * CDIM];
      acc = isMax ? fmaxf(acc, v) : acc + v;
    }
  if (!isMax) acc *= 1.f / (float)(k * k);
  long oidx = ((long)b * PTOK + tk) * CDIM + c;
  ptf[oidx] = acc;
  ptb[oidx] = (__bf16)acc;
}

// ---------------------------------------------------------------------------
// P2M softmax: fp32 scores row[KTPAD] -> bf16 probs written in place
// (bf16 row occupies the first half of the fp32 row; lda becomes 4096).
// ---------------------------------------------------------------------------
__global__ __launch_bounds__(256) void softmax_p2m(float* __restrict__ scores,
                                                   int Tvalid, float scale) {
  long row = blockIdx.x;
  float* rp = scores + row * KTPAD;
  __shared__ float red[256];
  int tid = threadIdx.x;
  float mx = -1e30f;
#pragma unroll
  for (int i = 0; i < 8; i++) {
    int t = tid + i * 256;
    if (t < Tvalid) mx = fmaxf(mx, rp[t]);
  }
  red[tid] = mx; __syncthreads();
  for (int s = 128; s > 0; s >>= 1) {
    if (tid < s) red[tid] = fmaxf(red[tid], red[tid + s]);
    __syncthreads();
  }
  mx = red[0]; __syncthreads();
  float pv[8];
  float sum = 0.f;
#pragma unroll
  for (int i = 0; i < 8; i++) {
    int t = tid + i * 256;
    float p = 0.f;
    if (t < Tvalid) p = __expf((rp[t] - mx) * scale);
    pv[i] = p; sum += p;
  }
  red[tid] = sum; __syncthreads();
  for (int s = 128; s > 0; s >>= 1) {
    if (tid < s) red[tid] += red[tid + s];
    __syncthreads();
  }
  float inv = 1.f / red[0];
  __bf16* op = (__bf16*)rp;   // all fp32 reads complete: barriers above
#pragma unroll
  for (int i = 0; i < 8; i++) {
    int t = tid + i * 256;
    op[t] = (__bf16)(pv[i] * inv);
  }
}

// ---------------------------------------------------------------------------
// bf16 transpose: in[b, t(R), c(Cc)] -> out[b, c, t], zero for t >= Tvalid.
// ---------------------------------------------------------------------------
__global__ void transpose_bf16(const __bf16* __restrict__ in,
                               __bf16* __restrict__ out, int R, int Cc,
                               int Tvalid, int total) {
  int i = blockIdx.x * 256 + threadIdx.x;
  if (i >= total) return;
  int t = i % R;
  int rest = i / R;
  int c = rest % Cc;
  int b = rest / Cc;
  out[i] = (t < Tvalid) ? in[((long)b * R + t) * Cc + c] : (__bf16)0.0f;
}

// Weight transpose: W[K,N] fp32 -> Wt[N,K] bf16
__global__ void wtrans_kernel(const float* __restrict__ W,
                              __bf16* __restrict__ Wt, int K, int N) {
  int i = blockIdx.x * 256 + threadIdx.x;
  if (i >= K * N) return;
  int k = i % K, n = i / K;
  Wt[i] = (__bf16)W[(long)k * N + n];
}

// mp output: pt region5 [B,900,256] -> out [B,256,900]
__global__ void mp_out_kernel(const float* __restrict__ pt,
                              float* __restrict__ out3) {
  int i = blockIdx.x * 256 + threadIdx.x;
  if (i >= BATCH * CDIM * 900) return;
  int t = i % 900;
  int rest = i / 900;
  int c = rest % CDIM;
  int b = rest / CDIM;
  out3[i] = pt[((long)b * PTOK + 869 + t) * CDIM + c];
}

// ---------------------------------------------------------------------------
// Fused M2P attention (T=256, d=32): one wave = 16 query rows for one (b,h).
// scores (16 WMMA) -> in-register softmax -> P via LDS -> attn@V (16 WMMA).
// Fully unrolled straight-line loops: compiler pipelines loads freely.
// ---------------------------------------------------------------------------
__global__ __launch_bounds__(128) void m2p_attn(
    const __bf16* __restrict__ q2, const __bf16* __restrict__ kv2,
    const __bf16* __restrict__ kv2t, __bf16* __restrict__ outp) {
  const int z = blockIdx.z;
  const int b = z >> 3, h = z & 7;
  const int wv = threadIdx.x >> 5;
  const int lane = threadIdx.x & 31;
  const int lh = lane >> 4, ll = lane & 15;
  const int qbase = blockIdx.x * 64 + wv * 16;

  __shared__ __bf16 Plds[4 * 16 * 256];

  int qRow = qbase + ll; if (qRow > NTOK - 1) qRow = NTOK - 1;
  const __bf16* qp = q2 + ((long)b * NTOK + qRow) * CDIM + h * DHEAD;
  v16bf qv = ld16(qp + lh * 16, qp + lh * 16 + 8);

  const __bf16* kbase = kv2 + (long)b * 256 * (2 * CDIM) + h * DHEAD;
  v8f sc[16];
  {
    const __bf16* kp0 = kbase + (long)ll * (2 * CDIM);
    v16bf kvf = ld16(kp0 + lh * 16, kp0 + lh * 16 + 8);
#pragma unroll
    for (int t = 0; t < 16; t++) {
      int tn = (t + 1 < 16) ? t + 1 : t;
      const __bf16* kp = kbase + (long)(tn * 16 + ll) * (2 * CDIM);
      v16bf kvn = ld16(kp + lh * 16, kp + lh * 16 + 8);
      v8f zero;
#pragma unroll
      for (int j = 0; j < 8; j++) zero[j] = 0.f;
      sc[t] = wmma_bf16(qv, kvf, zero);
      kvf = kvn;
    }
  }

  const float scale = 0.17677669529663687f;  // 1/sqrt(32)
#pragma unroll
  for (int j = 0; j < 8; j++) {
    float mx = -1e30f;
#pragma unroll
    for (int t = 0; t < 16; t++) mx = fmaxf(mx, sc[t][j]);
#pragma unroll
    for (int s = 1; s < 16; s <<= 1) mx = fmaxf(mx, __shfl_xor(mx, s, 32));
    float sum = 0.f;
#pragma unroll
    for (int t = 0; t < 16; t++) {
      float p = __expf((sc[t][j] - mx) * scale);
      sc[t][j] = p; sum += p;
    }
#pragma unroll
    for (int s = 1; s < 16; s <<= 1) sum += __shfl_xor(sum, s, 32);
    float inv = 1.f / sum;
#pragma unroll
    for (int t = 0; t < 16; t++) sc[t][j] *= inv;
  }

  __bf16* pw = Plds + wv * 16 * 256;
#pragma unroll
  for (int t = 0; t < 16; t++)
#pragma unroll
    for (int j = 0; j < 8; j++)
      pw[(lh * 8 + j) * 256 + t * 16 + ll] = (__bf16)sc[t][j];
  __syncthreads();

  v8f oa[2];
#pragma unroll
  for (int nt = 0; nt < 2; nt++)
#pragma unroll
    for (int j = 0; j < 8; j++) oa[nt][j] = 0.f;

  const __bf16* vbase =
      kv2t + ((long)b * (2 * CDIM) + CDIM + h * DHEAD) * 256;
  {
    const __bf16* pa0 = pw + ll * 256;
    v16bf av = ld16(pa0 + lh * 16, pa0 + lh * 16 + 8);
    v16bf bvv[2];
#pragma unroll
    for (int nt = 0; nt < 2; nt++) {
      const __bf16* vp = vbase + (long)(nt * 16 + ll) * 256;
      bvv[nt] = ld16(vp + lh * 16, vp + lh * 16 + 8);
    }
#pragma unroll
    for (int kk = 0; kk < 8; kk++) {
      int kn = (kk + 1 < 8) ? kk + 1 : kk;
      const __bf16* pan = pw + ll * 256 + kn * 32;
      v16bf avn = ld16(pan + lh * 16, pan + lh * 16 + 8);
      v16bf bvn[2];
#pragma unroll
      for (int nt = 0; nt < 2; nt++) {
        const __bf16* vp = vbase + (long)(nt * 16 + ll) * 256 + kn * 32;
        bvn[nt] = ld16(vp + lh * 16, vp + lh * 16 + 8);
      }
#pragma unroll
      for (int nt = 0; nt < 2; nt++) oa[nt] = wmma_bf16(av, bvv[nt], oa[nt]);
      av = avn;
#pragma unroll
      for (int nt = 0; nt < 2; nt++) bvv[nt] = bvn[nt];
    }
  }

#pragma unroll
  for (int nt = 0; nt < 2; nt++)
#pragma unroll
    for (int j = 0; j < 8; j++) {
      int row = qbase + lh * 8 + j;
      if (row < NTOK)
        outp[((long)b * NTOK + row) * CDIM + h * DHEAD + nt * 16 + ll] =
            (__bf16)oa[nt][j];
    }
}

// ---------------------------------------------------------------------------
// Host side
// ---------------------------------------------------------------------------
static void gemm(hipStream_t st, const void* A, long sAb, long sAh, int lda,
                 const void* Bt, long sBb, long sBh, int ldb, void* Cf, void* Cb,
                 long sCb, long sCh, int ldc, const void* bias, const void* res,
                 int M, int N, int K, int nbh, int nB, int doGelu) {
  dim3 g((N + 63) / 64, (M + 63) / 64, nB * nbh);
  gemm_nt_wmma<<<g, 128, 0, st>>>(
      (const __bf16*)A, sAb, sAh, lda, (const __bf16*)Bt, sBb, sBh, ldb,
      (float*)Cf, (__bf16*)Cb, sCb, sCh, ldc, (const float*)bias,
      (const float*)res, M, N, K, nbh, doGelu);
}

extern "C" void kernel_launch(void* const* d_in, const int* in_sizes, int n_in,
                              void* d_out, int out_size, void* d_ws,
                              size_t ws_size, hipStream_t stream) {
  (void)in_sizes; (void)n_in; (void)out_size; (void)ws_size;
  const float* x_in = (const float*)d_in[0];
  const float* m_in = (const float*)d_in[1];
  // params: n1..n6,np (w,b) = 2..15; c1..c5 (w,b) = 16..25;
  // p2m q/kv/proj_w/proj_b = 26..29; m2p = 30..33;
  // mx w1,b1,w2,b2 = 34..37; mm = 38..41
  float* dout = (float*)d_out;

  char* ws = (char*)d_ws;
  size_t cur = 0;
  auto alloc = [&](size_t bytes) -> char* {
    char* p = ws + cur;
    cur = (cur + bytes + 255) & ~(size_t)255;
    return p;
  };

  __bf16* wt_c[5];
  for (int i = 0; i < 5; i++) wt_c[i] = (__bf16*)alloc(65536 * 2);
  __bf16* wt_p2m_q    = (__bf16*)alloc(65536 * 2);
  __bf16* wt_p2m_kv   = (__bf16*)alloc(131072 * 2);
  __bf16* wt_p2m_proj = (__bf16*)alloc(65536 * 2);
  __bf16* wt_m2p_q    = (__bf16*)alloc(65536 * 2);
  __bf16* wt_m2p_kv   = (__bf16*)alloc(131072 * 2);
  __bf16* wt_m2p_proj = (__bf16*)alloc(65536 * 2);
  __bf16* wt_mx1 = (__bf16*)alloc(262144 * 2);
  __bf16* wt_mx2 = (__bf16*)alloc(262144 * 2);
  __bf16* wt_mm1 = (__bf16*)alloc(262144 * 2);
  __bf16* wt_mm2 = (__bf16*)alloc(262144 * 2);

  __bf16* xn_bf  = (__bf16*)alloc((size_t)BATCH * NTOK * CDIM * 2);   // reused ln1/ln3/ln6
  float*  pt     = (float*)alloc((size_t)BATCH * PTOK * CDIM * 4);    // pooled, conv'd in place
  __bf16* pt_bf  = (__bf16*)alloc((size_t)BATCH * PTOK * CDIM * 2);
  __bf16* kv_in  = (__bf16*)alloc((size_t)BATCH * KTVAL * CDIM * 2);
  __bf16* qbuf   = (__bf16*)alloc((size_t)BATCH * 256 * CDIM * 2);
  __bf16* kvbuf  = (__bf16*)alloc((size_t)BATCH * KTPAD * 512 * 2);
  float*  scores = (float*)alloc((size_t)BATCH * NHEAD * 256 * KTPAD * 4);
  __bf16* kvt    = (__bf16*)alloc((size_t)BATCH * 512 * KTPAD * 2);
  __bf16* attno  = (__bf16*)alloc((size_t)BATCH * 256 * CDIM * 2);
  float*  mbuf2  = (float*)alloc((size_t)BATCH * 256 * CDIM * 4);
  __bf16* m4_bf  = (__bf16*)alloc((size_t)BATCH * 256 * CDIM * 2);
  __bf16* m5_bf  = (__bf16*)alloc((size_t)BATCH * 256 * CDIM * 2);
  __bf16* q2     = (__bf16*)alloc((size_t)BATCH * NTOK * CDIM * 2);
  __bf16* kv2    = (__bf16*)alloc((size_t)BATCH * 256 * 512 * 2);
  __bf16* kv2t   = (__bf16*)alloc((size_t)BATCH * 512 * 256 * 2);
  __bf16* attno2 = (__bf16*)alloc((size_t)BATCH * NTOK * CDIM * 2);
  float*  xbuf2  = (float*)alloc((size_t)BATCH * NTOK * CDIM * 4);
  __bf16* hidm   = (__bf16*)alloc((size_t)BATCH * 256 * DFF * 2);
  __bf16* hidx   = (__bf16*)alloc((size_t)BATCH * NTOK * DFF * 2);

  auto wtr = [&](int idx, __bf16* dst, int K, int N) {
    int tot = K * N;
    wtrans_kernel<<<(tot + 255) / 256, 256, 0, stream>>>((const float*)d_in[idx],
                                                         dst, K, N);
  };
  auto ln = [&](const void* inp, int gw, int gb, void* outp, int totalRows,
                int rpb, long obs, long ooff) {
    layernorm_bf16<<<(totalRows + 3) / 4, 128, 0, stream>>>(
        (const float*)inp, (const float*)d_in[gw], (const float*)d_in[gb],
        (__bf16*)outp, totalRows, rpb, obs, ooff);
  };

  // --- weight transposes (fp32 -> bf16 [N,K]) ---
  for (int i = 0; i < 5; i++) wtr(16 + 2 * i, wt_c[i], 256, 256);
  wtr(26, wt_p2m_q, 256, 256);   wtr(27, wt_p2m_kv, 256, 512);
  wtr(28, wt_p2m_proj, 256, 256);
  wtr(30, wt_m2p_q, 256, 256);   wtr(31, wt_m2p_kv, 256, 512);
  wtr(32, wt_m2p_proj, 256, 256);
  wtr(34, wt_mx1, 256, 1024);    wtr(36, wt_mx2, 1024, 256);
  wtr(38, wt_mm1, 256, 1024);    wtr(40, wt_mm2, 1024, 256);

  const long SKV = (long)KTVAL * CDIM;  // kv_in batch stride

  // --- LN(x;n1) -> xn, pool, 1x1-conv residuals ---
  ln(x_in, 2, 3, xn_bf, BATCH * NTOK, BATCH * NTOK, 0, 0);
  pool_kernel<<<BATCH * PTOK, 256, 0, stream>>>(xn_bf, pt, pt_bf);
  {
    int roff[5] = {0, 400, 625, 769, 869};
    int rsz[5] = {400, 225, 144, 100, 900};
    for (int r = 0; r < 5; r++)
      gemm(stream, pt_bf + (long)roff[r] * CDIM, (long)PTOK * CDIM, 0, CDIM,
           wt_c[r], 0, 0, CDIM, pt + (long)roff[r] * CDIM, nullptr,
           (long)PTOK * CDIM, 0, CDIM, d_in[17 + 2 * r],
           pt + (long)roff[r] * CDIM, rsz[r], CDIM, CDIM, 1, BATCH, 0);
  }

  // --- kv_in = [ LN(pt;np) ; LN(m;n2) ] ---
  ln(m_in, 4, 5, kv_in, BATCH * 256, 256, SKV, (long)PTOK * CDIM);
  ln(pt, 14, 15, kv_in, BATCH * PTOK, PTOK, SKV, 0);

  // --- P2M projections ---
  gemm(stream, kv_in + (long)PTOK * CDIM, SKV, 0, CDIM, wt_p2m_q, 0, 0, CDIM,
       nullptr, qbuf, 65536, 0, CDIM, nullptr, nullptr, 256, CDIM, CDIM, 1,
       BATCH, 0);
  gemm(stream, kv_in, SKV, 0, CDIM, wt_p2m_kv, 0, 0, CDIM, nullptr, kvbuf,
       (long)KTPAD * 512, 0, 512, nullptr, nullptr, KTVAL, 512, CDIM, 1, BATCH,
       0);

  // --- P2M scores (per b,h): [256 x 2048] = Q[256x32] x K^T ---
  gemm(stream, qbuf, 65536, DHEAD, CDIM, kvbuf, (long)KTPAD * 512, DHEAD, 512,
       scores, nullptr, (long)NHEAD * 256 * KTPAD, (long)256 * KTPAD, KTPAD,
       nullptr, nullptr, 256, KTPAD, DHEAD, NHEAD, BATCH, 0);
  softmax_p2m<<<BATCH * NHEAD * 256, 256, 0, stream>>>(scores, KTVAL,
                                                       0.17677669529663687f);
  transpose_bf16<<<(BATCH * 512 * KTPAD + 255) / 256, 256, 0, stream>>>(
      kvbuf, kvt, KTPAD, 512, KTVAL, BATCH * 512 * KTPAD);

  // --- attn @ V (probs bf16 in-place over scores: lda=4096) ---
  gemm(stream, scores, (long)NHEAD * 256 * 4096, (long)256 * 4096, 4096,
       kvt + (long)CDIM * KTPAD, (long)512 * KTPAD, (long)DHEAD * KTPAD, KTPAD,
       nullptr, attno, 65536, DHEAD, CDIM, nullptr, nullptr, 256, DHEAD, KTPAD,
       NHEAD, BATCH, 0);

  // --- P2M proj + residual: m2 = m + attno @ proj ---
  gemm(stream, attno, 65536, 0, CDIM, wt_p2m_proj, 0, 0, CDIM, mbuf2, nullptr,
       65536, 0, CDIM, d_in[29], m_in, 256, CDIM, CDIM, 1, BATCH, 0);

  // --- M2P attention ---
  ln(mbuf2, 8, 9, m4_bf, BATCH * 256, BATCH * 256, 0, 0);   // n4
  ln(x_in, 6, 7, xn_bf, BATCH * NTOK, BATCH * NTOK, 0, 0);  // n3
  gemm(stream, xn_bf, (long)NTOK * CDIM, 0, CDIM, wt_m2p_q, 0, 0, CDIM, nullptr,
       q2, (long)NTOK * CDIM, 0, CDIM, nullptr, nullptr, NTOK, CDIM, CDIM, 1,
       BATCH, 0);
  gemm(stream, m4_bf, 65536, 0, CDIM, wt_m2p_kv, 0, 0, CDIM, nullptr, kv2,
       (long)256 * 512, 0, 512, nullptr, nullptr, 256, 512, CDIM, 1, BATCH, 0);
  transpose_bf16<<<(BATCH * 512 * 256 + 255) / 256, 256, 0, stream>>>(
      kv2, kv2t, 256, 512, 256, BATCH * 512 * 256);
  m2p_attn<<<dim3((NTOK + 63) / 64, 1, BATCH * NHEAD), 128, 0, stream>>>(
      q2, kv2, kv2t, attno2);
  gemm(stream, attno2, (long)NTOK * CDIM, 0, CDIM, wt_m2p_proj, 0, 0, CDIM,
       xbuf2, nullptr, (long)NTOK * CDIM, 0, CDIM, d_in[33], x_in, NTOK, CDIM,
       CDIM, 1, BATCH, 0);

  // --- m MLP: d_out[m] = m2 + W2(gelu(W1 ln5(m2))) ---
  ln(mbuf2, 10, 11, m5_bf, BATCH * 256, BATCH * 256, 0, 0);  // n5
  gemm(stream, m5_bf, 65536, 0, CDIM, wt_mm1, 0, 0, CDIM, nullptr, hidm,
       (long)256 * DFF, 0, DFF, d_in[39], nullptr, 256, DFF, CDIM, 1, BATCH, 1);
  gemm(stream, hidm, (long)256 * DFF, 0, DFF, wt_mm2, 0, 0, DFF,
       dout + (long)BATCH * NTOK * CDIM, nullptr, 65536, 0, CDIM, d_in[41],
       mbuf2, 256, CDIM, DFF, 1, BATCH, 0);

  // --- x MLP: d_out[x] = x2 + W2(gelu(W1 ln6(x2))) ---
  ln(xbuf2, 12, 13, xn_bf, BATCH * NTOK, BATCH * NTOK, 0, 0);  // n6
  gemm(stream, xn_bf, (long)NTOK * CDIM, 0, CDIM, wt_mx1, 0, 0, CDIM, nullptr,
       hidx, (long)NTOK * DFF, 0, DFF, d_in[35], nullptr, NTOK, DFF, CDIM, 1,
       BATCH, 1);
  gemm(stream, hidx, (long)NTOK * DFF, 0, DFF, wt_mx2, 0, 0, DFF, dout, nullptr,
       (long)NTOK * CDIM, 0, CDIM, d_in[37], xbuf2, NTOK, CDIM, DFF, 1, BATCH,
       0);

  // --- mp output [B,256,30,30] ---
  mp_out_kernel<<<(BATCH * CDIM * 900 + 255) / 256, 256, 0, stream>>>(
      pt, dout + (long)BATCH * NTOK * CDIM + (long)BATCH * 256 * CDIM);
}
